// TriangularAttention_33174327394919
// MI455X (gfx1250) — matrix operand order repaired and checked by reading, not compiled, not run on hardware
//
#include <hip/hip_runtime.h>
#include <hip/hip_bf16.h>

typedef __attribute__((ext_vector_type(16))) __bf16 v16bf;
typedef __attribute__((ext_vector_type(8)))  __bf16 v8bf;
typedef __attribute__((ext_vector_type(8)))  float  v8f;

#define BATCH 32
#define SEQ   1024
#define DIM   256
#define SCALE 0.0625f   // 256^-0.5

// ---------------------------------------------------------------------------
// Kernel 0: one-shot fp32 -> bf16 conversion (hs, Wq, Wk). 8 elements/thread.
// ---------------------------------------------------------------------------
__global__ __launch_bounds__(256) void cvt_bf16_kernel(
    const float* __restrict__ src, __bf16* __restrict__ dst)
{
    const size_t i = ((size_t)blockIdx.x * blockDim.x + threadIdx.x) * 8;
    const float4 f0 = *(const float4*)(src + i);
    const float4 f1 = *(const float4*)(src + i + 4);
    v8bf o;
    o[0] = (__bf16)f0.x; o[1] = (__bf16)f0.y;
    o[2] = (__bf16)f0.z; o[3] = (__bf16)f0.w;
    o[4] = (__bf16)f1.x; o[5] = (__bf16)f1.y;
    o[6] = (__bf16)f1.z; o[7] = (__bf16)f1.w;
    *(v8bf*)(dst + i) = o;
}

// A fragment (16x32 bf16): lane l<16 -> row l, K={k0..k0+15};
// upper half-wave -> K={k0+16..k0+31} (matches B fragment order).
__device__ __forceinline__ v16bf load_a_frag(const __bf16* __restrict__ rowp,
                                             int k0, int half)
{
    const v8bf lo = *(const v8bf*)(rowp + k0 + half * 16);
    const v8bf hi = *(const v8bf*)(rowp + k0 + half * 16 + 8);
    v16bf a;
    #pragma unroll
    for (int e = 0; e < 8; ++e) { a[e] = lo[e]; a[e + 8] = hi[e]; }
    return a;
}

// B fragment (32x16 bf16, K-major): lane n<16 -> column n, K={k0..k0+15};
// upper half-wave -> K={k0+16..k0+31}. Column n = contiguous row n of source.
__device__ __forceinline__ v16bf load_b_frag(const __bf16* __restrict__ rowp,
                                             int k0, int half)
{
    return *(const v16bf*)(rowp + k0 + half * 16);
}

// ---------------------------------------------------------------------------
// Kernel 1: Q/K projection from bf16 operands. One wave -> one 16x16 tile of
// BOTH q and k (shared A fragment). fp32 WMMA accumulate, bias in fp32,
// stored bf16.
// ---------------------------------------------------------------------------
__global__ __launch_bounds__(256) void qk_project_kernel(
    const __bf16* __restrict__ hsb,
    const __bf16* __restrict__ wqb, const float* __restrict__ bq,
    const __bf16* __restrict__ wkb, const float* __restrict__ bk,
    __bf16* __restrict__ qbf, __bf16* __restrict__ kbf)
{
    const int wave = (blockIdx.x * blockDim.x + threadIdx.x) >> 5;
    const int lane = threadIdx.x & 31;
    const int half = lane >> 4;
    const int l16  = lane & 15;

    const int rowBase = (wave >> 4) << 4;   // 2048 row tiles of B*K=32768
    const int colBase = (wave & 15) << 4;   // 16 col tiles of D=256

    const __bf16* aRowP  = hsb + (size_t)(rowBase + l16) * DIM;
    const __bf16* bqRowP = wqb + (size_t)(colBase + l16) * DIM;  // x @ W^T
    const __bf16* bkRowP = wkb + (size_t)(colBase + l16) * DIM;

    v8f accQ = {};
    v8f accK = {};
    #pragma unroll
    for (int k0 = 0; k0 < DIM; k0 += 32) {
        const v16bf a   = load_a_frag(aRowP, k0, half);
        const v16bf bbq = load_b_frag(bqRowP, k0, half);
        const v16bf bbk = load_b_frag(bkRowP, k0, half);
        accQ = __builtin_amdgcn_wmma_f32_16x16x32_bf16(
            false, a, false, bbq, (short)0, accQ, false, false);
        accK = __builtin_amdgcn_wmma_f32_16x16x32_bf16(
            false, a, false, bbk, (short)0, accK, false, false);
    }

    // C/D layout: element r -> row (r + half*8), column (lane&15).
    const int col  = colBase + l16;
    const float vq = bq[col];
    const float vk = bk[col];
    #pragma unroll
    for (int r = 0; r < 8; ++r) {
        const int row = rowBase + r + half * 8;
        qbf[(size_t)row * DIM + col] = (__bf16)(accQ[r] + vq);
        kbf[(size_t)row * DIM + col] = (__bf16)(accK[r] + vk);
    }
}

// ---------------------------------------------------------------------------
// Kernel 2: scores = tril((f(q @ k^T * SCALE) + 0.001) * mask_i * mask_j),
// f = softplus on the diagonal, identity elsewhere.
// One wave -> 16x64 output strip (1 itile x 4 jtiles): A fragment reused 4x,
// 32 WMMAs per wave. Fully-upper strips zero-fill only.
// ---------------------------------------------------------------------------
__device__ __forceinline__ float softplusf(float x) {
    if (x > 20.0f) return x;
    return log1pf(__expf(x));
}

__global__ __launch_bounds__(256) void tri_scores_kernel(
    const __bf16* __restrict__ qbf, const __bf16* __restrict__ kbf,
    const float* __restrict__ mask, float* __restrict__ out)
{
    const int wave = (blockIdx.x * blockDim.x + threadIdx.x) >> 5;
    const int lane = threadIdx.x & 31;
    const int half = lane >> 4;
    const int l16  = lane & 15;

    const int b     = wave >> 10;        // 64 itiles * 16 jgroups per batch
    const int rem   = wave & 1023;
    const int itile = rem >> 4;          // 0..63
    const int jg    = rem & 15;          // 0..15 (4 jtiles each)
    const int iBase = itile << 4;
    const int jBase = jg << 6;           // 64-wide strip

    float* outB = out + (size_t)b * SEQ * SEQ;

    if ((jg << 2) > itile) {             // entire strip strictly above diagonal
        #pragma unroll
        for (int r = 0; r < 8; ++r) {
            const int i = iBase + r + half * 8;
            float4 z = {0.f, 0.f, 0.f, 0.f};
            // 4 tiles: columns jBase + t*16 + l16
            #pragma unroll
            for (int t = 0; t < 4; ++t)
                outB[(size_t)i * SEQ + jBase + t * 16 + l16] = 0.0f;
            (void)z;
        }
        return;
    }

    const __bf16* qRowP = qbf + (size_t)b * SEQ * DIM + (size_t)(iBase + l16) * DIM;
    const __bf16* kRowP = kbf + (size_t)b * SEQ * DIM + (size_t)(jBase + l16) * DIM;

    v8f acc0 = {}, acc1 = {}, acc2 = {}, acc3 = {};
    #pragma unroll
    for (int k0 = 0; k0 < DIM; k0 += 32) {
        const v16bf a  = load_a_frag(qRowP, k0, half);
        const v16bf b0 = load_b_frag(kRowP,                 k0, half);
        const v16bf b1 = load_b_frag(kRowP + 16 * DIM,      k0, half);
        const v16bf b2 = load_b_frag(kRowP + 32 * DIM,      k0, half);
        const v16bf b3 = load_b_frag(kRowP + 48 * DIM,      k0, half);
        acc0 = __builtin_amdgcn_wmma_f32_16x16x32_bf16(false, a, false, b0, (short)0, acc0, false, false);
        acc1 = __builtin_amdgcn_wmma_f32_16x16x32_bf16(false, a, false, b1, (short)0, acc1, false, false);
        acc2 = __builtin_amdgcn_wmma_f32_16x16x32_bf16(false, a, false, b2, (short)0, acc2, false, false);
        acc3 = __builtin_amdgcn_wmma_f32_16x16x32_bf16(false, a, false, b3, (short)0, acc3, false, false);
    }

    float mj[4];
    #pragma unroll
    for (int t = 0; t < 4; ++t) mj[t] = mask[b * SEQ + jBase + t * 16 + l16];

    #pragma unroll
    for (int r = 0; r < 8; ++r) {
        const int i = iBase + r + half * 8;
        const float mi = mask[b * SEQ + i];
        #pragma unroll
        for (int t = 0; t < 4; ++t) {
            const int j = jBase + t * 16 + l16;
            const float accv = (t == 0) ? acc0[r] : (t == 1) ? acc1[r]
                             : (t == 2) ? acc2[r] : acc3[r];
            float v = 0.0f;
            if (j <= i) {
                const float s = accv * SCALE;
                const float combined = (i == j) ? softplusf(s) : s;
                v = (combined + 0.001f) * mi * mj[t];
            }
            outB[(size_t)i * SEQ + j] = v;
        }
    }
}

// ---------------------------------------------------------------------------
extern "C" void kernel_launch(void* const* d_in, const int* in_sizes, int n_in,
                              void* d_out, int out_size, void* d_ws, size_t ws_size,
                              hipStream_t stream) {
    const float* hs   = (const float*)d_in[0];   // [B,K,D]
    const float* mask = (const float*)d_in[1];   // [B,K]
    const float* Wq   = (const float*)d_in[2];   // [D,D]
    const float* bq   = (const float*)d_in[3];   // [D]
    const float* Wk   = (const float*)d_in[4];   // [D,D]
    const float* bk   = (const float*)d_in[5];   // [D]
    float* out = (float*)d_out;                  // [B,K,K]

    const size_t HS_N = (size_t)BATCH * SEQ * DIM;   // 8388608
    const size_t W_N  = (size_t)DIM * DIM;           // 65536

    __bf16* hsb = (__bf16*)d_ws;
    __bf16* wqb = hsb + HS_N;
    __bf16* wkb = wqb + W_N;
    __bf16* qbf = wkb + W_N;
    __bf16* kbf = qbf + HS_N;                        // total ~50.6 MB

    // One-shot conversions (8 elems/thread).
    cvt_bf16_kernel<<<(int)(HS_N / 8 / 256), 256, 0, stream>>>(hs, hsb);
    cvt_bf16_kernel<<<(int)(W_N  / 8 / 256), 256, 0, stream>>>(Wq, wqb);
    cvt_bf16_kernel<<<(int)(W_N  / 8 / 256), 256, 0, stream>>>(Wk, wkb);

    // Projection: 2048*16 wave-tiles / 8 waves per block.
    qk_project_kernel<<<4096, 256, 0, stream>>>(hsb, wqb, bq, wkb, bk, qbf, kbf);

    // Scores: 32 batches * 64 itiles * 16 jgroups = 32768 waves / 8 per block.
    tri_scores_kernel<<<4096, 256, 0, stream>>>(qbf, kbf, mask, out);
}
